// MHA_1692217115250
// MI455X (gfx1250) — compile-verified, matrix operand order checked by reading, not run on hardware
//
#include <hip/hip_runtime.h>
#include <hip/hip_bf16.h>
#include <cstdint>

typedef __attribute__((ext_vector_type(16))) _Float16 v16h;
typedef __attribute__((ext_vector_type(8)))  _Float16 v8h;
typedef __attribute__((ext_vector_type(4)))  _Float16 v4h;
typedef __attribute__((ext_vector_type(8)))  float    v8f;
typedef __attribute__((ext_vector_type(4)))  float    v4f;
typedef __attribute__((ext_vector_type(4)))  unsigned int u32x4;
typedef __attribute__((ext_vector_type(8)))  int      i32x8;
typedef __attribute__((ext_vector_type(4)))  int      i32x4;

// ---------------------------------------------------------------------------
// CDNA5 async / TDM helpers
// ---------------------------------------------------------------------------
__device__ __forceinline__ void async_ld_b128(void* lds, const void* g) {
#if __has_builtin(__builtin_amdgcn_global_load_async_to_lds_b128)
    __builtin_amdgcn_global_load_async_to_lds_b128(
        (__attribute__((address_space(1))) i32x4*)(uintptr_t)g,
        (__attribute__((address_space(3))) i32x4*)(uint32_t)(uintptr_t)lds,
        0, 0);
#else
    uint32_t l = (uint32_t)(uintptr_t)lds;
    unsigned long long ga = (unsigned long long)(uintptr_t)g;
    asm volatile("global_load_async_to_lds_b128 %0, %1, off"
                 :: "v"(l), "v"(ga) : "memory");
#endif
}

__device__ __forceinline__ void wait_async0() {
#if __has_builtin(__builtin_amdgcn_s_wait_asynccnt)
    __builtin_amdgcn_s_wait_asynccnt(0);
#else
    asm volatile("s_wait_asynccnt 0" ::: "memory");
#endif
}

__device__ __forceinline__ void wait_tensor0() {
#if __has_builtin(__builtin_amdgcn_s_wait_tensorcnt)
    __builtin_amdgcn_s_wait_tensorcnt(0);
#else
    asm volatile("s_wait_tensorcnt 0" ::: "memory");
#endif
}

// 2D TDM load: rows x rowlen (halves), global row stride in halves.
__device__ __forceinline__ void tdm_load_2d(_Float16* lds, const _Float16* g,
                                            int rows, int rowlen, int stride) {
#if __has_builtin(__builtin_amdgcn_tensor_load_to_lds)
    const uint64_t ga = (uint64_t)(uintptr_t)g;
    u32x4 g0;
    g0[0] = 1u;                                               // count=1, user desc
    g0[1] = (uint32_t)(uintptr_t)lds;                         // lds_addr
    g0[2] = (uint32_t)ga;                                     // global_addr[31:0]
    g0[3] = (uint32_t)((ga >> 32) & 0x1FFFFFFu) | (2u << 30); // addr[56:32] | type=2
    i32x8 g1;
    g1[0] = 0x10000;                                          // data_size=1 (2 bytes)
    g1[1] = (rowlen & 0xFFFF) << 16;                          // tensor_dim0[15:0]
    g1[2] = (rowlen >> 16) | ((rows & 0xFFFF) << 16);         // dim0 hi | dim1 lo
    g1[3] = (rows >> 16) | (rowlen << 16);                    // dim1 hi | tile_dim0
    g1[4] = rows;                                             // tile_dim1 (tile_dim2=0)
    g1[5] = stride;                                           // tensor_dim0_stride lo
    g1[6] = 0;
    g1[7] = 0;
    i32x4 z4 = {0, 0, 0, 0};
    i32x8 z8 = {0, 0, 0, 0, 0, 0, 0, 0};
    __builtin_amdgcn_tensor_load_to_lds(g0, g1, z4, z4, z8, 0);
#else
    const int lane = threadIdx.x & 31;
    const int total = rows * rowlen;
    for (int e = lane * 8; e < total; e += 32 * 8) {
        const int r = e / rowlen, c = e % rowlen;
        *(v8h*)&lds[e] = *(const v8h*)(g + (size_t)r * stride + c);
    }
#endif
}

// ---------------------------------------------------------------------------
// WMMA fragment helpers (wave32, v_wmma_f32_16x16x32_f16)
// ---------------------------------------------------------------------------
__device__ __forceinline__ v16h load_frag_rm(const _Float16* __restrict__ base, int ld) {
    const int lane = threadIdx.x & 31;
    const int idx  = lane & 15;
    const int hf   = lane >> 4;
    const _Float16* p = base + (size_t)idx * ld + hf * 8;
    v8h lo = *(const v8h*)(p);
    v8h hi = *(const v8h*)(p + 16);
    v16h r;
#pragma unroll
    for (int i = 0; i < 8; ++i) { r[i] = lo[i]; r[i + 8] = hi[i]; }
    return r;
}

__device__ __forceinline__ v8f wmma_f16(v16h a, v16h b, v8f c) {
    return __builtin_amdgcn_wmma_f32_16x16x32_f16(false, a, false, b, (short)0, c,
                                                  false, false);
}

// ---------------------------------------------------------------------------
// fp32 -> f16 conversion (x4 vectorized)
// ---------------------------------------------------------------------------
__global__ void cvt_f32_to_f16(const float* __restrict__ in, _Float16* __restrict__ out,
                               int n4) {
    int i = blockIdx.x * blockDim.x + threadIdx.x;
    if (i >= n4) return;
    v4f v = ((const v4f*)in)[i];
    v4h h;
#pragma unroll
    for (int j = 0; j < 4; ++j) h[j] = (_Float16)v[j];
    ((v4h*)out)[i] = h;
}

// ---------------------------------------------------------------------------
// GEMM: C[M,N] = A[M,K] * W[N,K]^T
// block 128 = 4 waves (2x2), wave tile 64x64, block tile 128x128, BK=32.
// 16 independent WMMAs per K-step vs 16 ds_load_b128 (ratio 1.0).
// Double-buffered LDS staging via ASYNC global->LDS b128 copies.
// ---------------------------------------------------------------------------
template <bool OUT16>
__global__ __launch_bounds__(128) void gemm_xwT(const _Float16* __restrict__ A,
                                                const _Float16* __restrict__ W,
                                                void* __restrict__ C,
                                                int N, int Kd) {
    __shared__ _Float16 As[2][128 * 32];   // 8KB each
    __shared__ _Float16 Ws[2][128 * 32];   // 8KB each

    const int tid  = threadIdx.x;
    const int wv   = tid >> 5;
    const int lane = tid & 31;
    const int idx  = lane & 15;
    const int hf   = lane >> 4;
    const int wm   = wv >> 1;              // 0..1
    const int wn   = wv & 1;               // 0..1
    const int mB   = blockIdx.y * 128;
    const int nB   = blockIdx.x * 128;

    v8f acc[4][4];
#pragma unroll
    for (int i = 0; i < 4; ++i)
#pragma unroll
        for (int j = 0; j < 4; ++j)
#pragma unroll
            for (int r = 0; r < 8; ++r) acc[i][j][r] = 0.0f;

    const int nk = Kd / 32;

    auto stage = [&](int buf, int k0) {
#pragma unroll
        for (int t = 0; t < 4; ++t) {
            const int c   = tid + t * 128;
            const int row = c >> 2, seg = c & 3;
            async_ld_b128(&As[buf][row * 32 + seg * 8],
                          A + (size_t)(mB + row) * Kd + k0 + seg * 8);
        }
#pragma unroll
        for (int t = 0; t < 4; ++t) {
            const int c   = tid + t * 128;
            const int row = c >> 2, seg = c & 3;
            async_ld_b128(&Ws[buf][row * 32 + seg * 8],
                          W + (size_t)(nB + row) * Kd + k0 + seg * 8);
        }
    };

    stage(0, 0);
    for (int kt = 0; kt < nk; ++kt) {
        wait_async0();
        __syncthreads();                       // tile kt ready; prev readers done
        if (kt + 1 < nk) stage((kt + 1) & 1, (kt + 1) * 32);

        const _Float16* a  = As[kt & 1];
        const _Float16* wb = Ws[kt & 1];
        v16h af[4], bf[4];
#pragma unroll
        for (int i = 0; i < 4; ++i)
            af[i] = load_frag_rm(a + (wm * 64 + i * 16) * 32, 32);
#pragma unroll
        for (int j = 0; j < 4; ++j)
            bf[j] = load_frag_rm(wb + (wn * 64 + j * 16) * 32, 32);
#pragma unroll
        for (int i = 0; i < 4; ++i)
#pragma unroll
            for (int j = 0; j < 4; ++j)
                acc[i][j] = wmma_f16(af[i], bf[j], acc[i][j]);
    }

#pragma unroll
    for (int i = 0; i < 4; ++i)
#pragma unroll
        for (int j = 0; j < 4; ++j)
#pragma unroll
            for (int r = 0; r < 8; ++r) {
                const int row = mB + wm * 64 + i * 16 + r + 8 * hf;
                const int col = nB + wn * 64 + j * 16 + idx;
                if (OUT16)
                    ((_Float16*)C)[(size_t)row * N + col] = (_Float16)acc[i][j][r];
                else
                    ((float*)C)[(size_t)row * N + col] = acc[i][j][r];
            }
}

// ---------------------------------------------------------------------------
// RoPE in place on [TOK, H*128] f16, scale folded in (1/sqrt(hd) for Q)
// ---------------------------------------------------------------------------
__global__ void rope_kernel(_Float16* __restrict__ X, float scale, int npairs) {
    int p = blockIdx.x * blockDim.x + threadIdx.x;
    if (p >= npairs) return;
    const int t   = p >> 10;
    const int rem = p & 1023;
    const int h   = rem >> 6;
    const int i   = rem & 63;
    const int s   = t & 2047;
    const float inv = __powf(10000.0f, -(float)i * (1.0f / 64.0f));
    const float fr  = (float)s * inv;
    float sn, cs;
    __sincosf(fr, &sn, &cs);
    const size_t base = (size_t)t * 2048 + (size_t)h * 128;
    const float x1 = (float)X[base + i];
    const float x2 = (float)X[base + i + 64];
    X[base + i]      = (_Float16)((x1 * cs - x2 * sn) * scale);
    X[base + i + 64] = (_Float16)((x2 * cs + x1 * sn) * scale);
}

// ---------------------------------------------------------------------------
// Flash attention: grid (S/64, H, B), block 128 (4 waves); Br=Bc=64, hd=128.
// Software-pipelined: TDM K-load + V global loads for tile kt+1 are issued
// before computing tile kt (double-buffered Ks/Vt). Causal masking only on
// the diagonal tile (kt == qt); interior tiles skip mask VALU entirely.
// ---------------------------------------------------------------------------
__global__ __launch_bounds__(128) void flash_attn_kernel(
    const _Float16* __restrict__ Q, const _Float16* __restrict__ K,
    const _Float16* __restrict__ V, _Float16* __restrict__ O) {
    constexpr int S = 2048, Dm = 2048, HD = 128, BC = 64;
    const int qt   = blockIdx.x;
    const int h    = blockIdx.y;
    const int b    = blockIdx.z;
    const int tid  = threadIdx.x;
    const int w    = tid >> 5;
    const int lane = tid & 31;
    const int idx  = lane & 15;
    const int hf   = lane >> 4;

    __shared__ _Float16 Ks[2][BC * 128];   // [key][feat]  (TDM destination)
    __shared__ _Float16 Vt[2][128 * BC];   // [feat][key]
    __shared__ _Float16 Ps[4][16 * BC];    // per-wave P re-layout buffer

    const size_t tok0 = (size_t)b * S;
    const _Float16* Qg = Q + tok0 * Dm + (size_t)h * HD;
    const _Float16* Kg = K + tok0 * Dm + (size_t)h * HD;
    const _Float16* Vg = V + tok0 * Dm + (size_t)h * HD;

    const int q0    = qt * 64;
    const int qrow0 = q0 + w * 16;

    v16h aq[4];
#pragma unroll
    for (int kk = 0; kk < 4; ++kk)
        aq[kk] = load_frag_rm(Qg + (size_t)qrow0 * Dm + kk * 32, Dm);

    v8f o[8];
#pragma unroll
    for (int c = 0; c < 8; ++c)
#pragma unroll
        for (int r = 0; r < 8; ++r) o[c][r] = 0.0f;
    float mrow[8], lrow[8];
#pragma unroll
    for (int r = 0; r < 8; ++r) { mrow[r] = -__builtin_inff(); lrow[r] = 0.0f; }

    v8h vbuf[8];
    auto loadV = [&](int kb) {
#pragma unroll
        for (int it = 0; it < 8; ++it) {
            const int e = it * 1024 + tid * 8;
            const int key = e >> 7, feat = e & 127;
            vbuf[it] = *(const v8h*)(Vg + (size_t)(kb + key) * Dm + feat);
        }
    };
    auto storeV = [&](int buf) {
#pragma unroll
        for (int it = 0; it < 8; ++it) {
            const int e = it * 1024 + tid * 8;
            const int key = e >> 7, feat = e & 127;
#pragma unroll
            for (int j = 0; j < 8; ++j) Vt[buf][(feat + j) * BC + key] = vbuf[it][j];
        }
    };

    // prologue: stage tile 0
    if (w == 0) tdm_load_2d(Ks[0], Kg, BC, 128, Dm);
    loadV(0);
    storeV(0);

    for (int kt = 0; kt <= qt; ++kt) {
        const int kbase = kt * BC;
        const int cur = kt & 1, nxt = cur ^ 1;

        if (w == 0) wait_tensor0();    // TDM for tile kt complete
        __syncthreads();               // Ks[cur]/Vt[cur] ready; buf[nxt] readers done

        // issue next tile's data movement before computing this one
        if (kt < qt) {
            if (w == 0) tdm_load_2d(Ks[nxt], Kg + (size_t)(kbase + BC) * Dm, BC, 128, Dm);
            loadV(kbase + BC);
        }

        // S = Q_tile(16x128) @ K_tile^T(128x64)
        v8f sf[4];
#pragma unroll
        for (int n = 0; n < 4; ++n) {
            v8f s;
#pragma unroll
            for (int r = 0; r < 8; ++r) s[r] = 0.0f;
#pragma unroll
            for (int kk = 0; kk < 4; ++kk) {
                v16h bk = load_frag_rm(&Ks[cur][(n * 16) * 128 + kk * 32], 128);
                s = wmma_f16(aq[kk], bk, s);
            }
            sf[n] = s;
        }

        // causal mask: only the diagonal tile needs it (wave-uniform branch)
        if (kt == qt) {
#pragma unroll
            for (int n = 0; n < 4; ++n) {
                const int kcol = kbase + n * 16 + idx;
#pragma unroll
                for (int r = 0; r < 8; ++r) {
                    const int qr = qrow0 + r + 8 * hf;
                    if (kcol > qr) sf[n][r] = -__builtin_inff();
                }
            }
        }

        // online softmax (rows live across the 16 lanes of a half-wave)
        float alpha[8];
#pragma unroll
        for (int r = 0; r < 8; ++r) {
            float m = sf[0][r];
#pragma unroll
            for (int n = 1; n < 4; ++n) m = fmaxf(m, sf[n][r]);
#pragma unroll
            for (int st = 1; st < 16; st <<= 1) m = fmaxf(m, __shfl_xor(m, st, 32));
            const float mn = fmaxf(mrow[r], m);
            alpha[r] = __expf(mrow[r] - mn);
            mrow[r]  = mn;
        }

        float rsum[8];
#pragma unroll
        for (int r = 0; r < 8; ++r) rsum[r] = 0.0f;
#pragma unroll
        for (int n = 0; n < 4; ++n)
#pragma unroll
            for (int r = 0; r < 8; ++r) {
                const float p = __expf(sf[n][r] - mrow[r]);
                rsum[r] += p;
                Ps[w][(r + 8 * hf) * BC + n * 16 + idx] = (_Float16)p;
            }
#pragma unroll
        for (int r = 0; r < 8; ++r) {
            float t = rsum[r];
#pragma unroll
            for (int st = 1; st < 16; st <<= 1) t += __shfl_xor(t, st, 32);
            lrow[r] = lrow[r] * alpha[r] + t;
        }

#pragma unroll
        for (int c = 0; c < 8; ++c)
#pragma unroll
            for (int r = 0; r < 8; ++r) o[c][r] *= alpha[r];

        __syncthreads();  // Ps visible before PV phase

        // O += P(16x64) @ V(64x128)
        v16h pa0 = load_frag_rm(&Ps[w][0], BC);
        v16h pa1 = load_frag_rm(&Ps[w][32], BC);
#pragma unroll
        for (int c = 0; c < 8; ++c) {
            v16h bv0 = load_frag_rm(&Vt[cur][(c * 16) * BC], BC);
            v16h bv1 = load_frag_rm(&Vt[cur][(c * 16) * BC + 32], BC);
            o[c] = wmma_f16(pa0, bv0, o[c]);
            o[c] = wmma_f16(pa1, bv1, o[c]);
        }

        // finish pipelining: commit next V tile to LDS (visible at next barrier)
        if (kt < qt) storeV(nxt);
    }

    float linv[8];
#pragma unroll
    for (int r = 0; r < 8; ++r) linv[r] = 1.0f / lrow[r];
#pragma unroll
    for (int c = 0; c < 8; ++c)
#pragma unroll
        for (int r = 0; r < 8; ++r) {
            const int row = qrow0 + r + 8 * hf;
            const int col = h * HD + c * 16 + idx;
            O[(tok0 + row) * Dm + col] = (_Float16)(o[c][r] * linv[r]);
        }
}

// ---------------------------------------------------------------------------
extern "C" void kernel_launch(void* const* d_in, const int* in_sizes, int n_in,
                              void* d_out, int out_size, void* d_ws, size_t ws_size,
                              hipStream_t stream) {
    (void)in_sizes; (void)n_in; (void)out_size; (void)ws_size;
    const float* x  = (const float*)d_in[0];
    const float* Wq = (const float*)d_in[1];
    const float* Wk = (const float*)d_in[2];
    const float* Wv = (const float*)d_in[3];
    const float* Wo = (const float*)d_in[4];
    float* out = (float*)d_out;

    constexpr int B = 4, S = 2048, D = 2048, H = 16;
    constexpr size_t TOK = (size_t)B * S;        // 8192
    constexpr size_t XN  = TOK * D;              // 16,777,216
    constexpr size_t WN  = (size_t)D * D;        // 4,194,304

    _Float16* Xh  = (_Float16*)d_ws;
    _Float16* Wqh = Xh + XN;
    _Float16* Wkh = Wqh + WN;
    _Float16* Wvh = Wkh + WN;
    _Float16* Woh = Wvh + WN;
    _Float16* Qh  = Woh + WN;
    _Float16* Kh  = Qh + XN;
    _Float16* Vh  = Kh + XN;
    _Float16* Ah  = Xh;  // reuse x-f16 buffer for attention output

    cvt_f32_to_f16<<<(int)(XN / 4 / 256), 256, 0, stream>>>(x, Xh, (int)(XN / 4));
    cvt_f32_to_f16<<<(int)(WN / 4 / 256), 256, 0, stream>>>(Wq, Wqh, (int)(WN / 4));
    cvt_f32_to_f16<<<(int)(WN / 4 / 256), 256, 0, stream>>>(Wk, Wkh, (int)(WN / 4));
    cvt_f32_to_f16<<<(int)(WN / 4 / 256), 256, 0, stream>>>(Wv, Wvh, (int)(WN / 4));
    cvt_f32_to_f16<<<(int)(WN / 4 / 256), 256, 0, stream>>>(Wo, Woh, (int)(WN / 4));

    dim3 gg(D / 128, (unsigned)(TOK / 128));     // (16, 64)
    gemm_xwT<true><<<gg, 128, 0, stream>>>(Xh, Wqh, Qh, D, D);
    gemm_xwT<true><<<gg, 128, 0, stream>>>(Xh, Wkh, Kh, D, D);
    gemm_xwT<true><<<gg, 128, 0, stream>>>(Xh, Wvh, Vh, D, D);

    const int npairs = (int)(TOK * H * 64);      // 8,388,608
    rope_kernel<<<npairs / 256, 256, 0, stream>>>(Qh, 0.08838834764831845f, npairs);
    rope_kernel<<<npairs / 256, 256, 0, stream>>>(Kh, 1.0f, npairs);

    flash_attn_kernel<<<dim3(S / 64, H, B), 128, 0, stream>>>(Qh, Kh, Vh, Ah);

    gemm_xwT<false><<<gg, 128, 0, stream>>>(Ah, Woh, out, D, D);
}
